// MSHISST_30571577213041
// MI455X (gfx1250) — matrix operand order repaired and checked by reading, not compiled, not run on hardware
//
#include <hip/hip_runtime.h>
#include <math.h>

// ---------------------------------------------------------------------------
// HAT OCAB block, fused for MI455X (gfx1250, wave32, WMMA 16x16x32 bf16).
// Compute-bound (~19.5 GFLOP vs ~35MB HBM traffic) -> all GEMM-shaped work
// through v_wmma_f32_16x16x32_bf16 (fp32 accum). Tile staging uses CDNA5
// async-to-LDS (ASYNCcnt) with double buffering so global latency overlaps
// the WMMA stream instead of stalling on s_wait_loadcnt.
// ---------------------------------------------------------------------------

typedef unsigned short u16;
typedef __attribute__((ext_vector_type(16))) __bf16         v16bf;
typedef __attribute__((ext_vector_type(16))) unsigned short v16u;
typedef __attribute__((ext_vector_type(8)))  float          v8f;

#define BATCH 4
#define HW    64
#define L     4096            // 64*64
#define M_ROWS 16384          // BATCH * L
#define C_    180
#define CP    192             // C padded to 6*32
#define NH_   6
#define D_    30
#define DP    32              // head dim padded
#define WSZ   16
#define OWSZ  24
#define NQ    256
#define NKV   576
#define C4    720
#define C4P   736             // 23*32

__device__ __forceinline__ u16 f2bf(float f) {
  unsigned int u = __float_as_uint(f);
  unsigned int r = (u + 0x7FFFu + ((u >> 16) & 1u)) >> 16;   // RNE
  return (u16)r;
}

__device__ __forceinline__ v8f wmma_bf16(v16bf a, v16bf b, v8f c) {
  return __builtin_amdgcn_wmma_f32_16x16x32_bf16(false, a, false, b,
                                                 (short)0, c, false, false);
}

// CDNA5 async copy global->LDS (tracked by ASYNCcnt, bypasses VGPRs).
// Low 32 bits of a flat shared-aperture address are the LDS byte offset.
__device__ __forceinline__ void async_cp_b128(void* lds, const void* g) {
  unsigned l = (unsigned)(unsigned long long)lds;
  asm volatile("global_load_async_to_lds_b128 %0, %1, off"
               :: "v"(l), "v"(g) : "memory");
}
__device__ __forceinline__ void wait_async0() {
  asm volatile("s_wait_asynccnt 0x0" ::: "memory");
}

// ---------------------------------------------------------------------------
// LayerNorm (fp32 in, C=180) -> bf16 out padded to CP=192 (pads zeroed).
// One wave per row (wave32).
// ---------------------------------------------------------------------------
__global__ __launch_bounds__(256) void ln_cvt_kernel(
    const float* __restrict__ x, const float* __restrict__ g,
    const float* __restrict__ bta, u16* __restrict__ out, int M) {
  int row  = blockIdx.x * 8 + (threadIdx.x >> 5);
  int lane = threadIdx.x & 31;
  if (row >= M) return;
  const float* xr = x + (size_t)row * C_;
  float v[6];
  float s = 0.f, s2 = 0.f;
  for (int i = 0; i < 6; ++i) {
    int j = lane + i * 32;
    float val = (j < C_) ? xr[j] : 0.f;
    v[i] = val; s += val; s2 += val * val;
  }
  for (int off = 16; off >= 1; off >>= 1) {
    s  += __shfl_xor(s,  off, 32);
    s2 += __shfl_xor(s2, off, 32);
  }
  float mean = s * (1.f / C_);
  float var  = s2 * (1.f / C_) - mean * mean;
  float inv  = rsqrtf(var + 1e-5f);
  u16* orow = out + (size_t)row * CP;
  for (int i = 0; i < 6; ++i) {
    int j = lane + i * 32;
    float r = (j < C_) ? (v[i] - mean) * inv * g[j] + bta[j] : 0.f;
    orow[j] = f2bf(r);     // covers j in [0,192): pads written as 0
  }
}

// ---------------------------------------------------------------------------
// fp32 [N,K] weight -> bf16 [N,Kp] (zero-padded along K).
// ---------------------------------------------------------------------------
__global__ __launch_bounds__(256) void cvt_w_kernel(
    const float* __restrict__ src, u16* __restrict__ dst, int N, int K, int Kp) {
  int idx = blockIdx.x * 256 + threadIdx.x;
  if (idx >= N * Kp) return;
  int n = idx / Kp, k = idx - n * Kp;
  float v = (k < K) ? src[(size_t)n * K + k] : 0.f;
  dst[idx] = f2bf(v);
}

// ---------------------------------------------------------------------------
// Expand relative-position bias: rpb[1521,6] gathered by rpi[256,576]
// -> biasmat [6][256][576] f32 (one gather total instead of per window).
// ---------------------------------------------------------------------------
__global__ __launch_bounds__(256) void bias_expand_kernel(
    const float* __restrict__ rpb, const int* __restrict__ rpi,
    float* __restrict__ bm) {
  int idx = blockIdx.x * 256 + threadIdx.x;      // over NQ*NKV
  if (idx >= NQ * NKV) return;
  int r = rpi[idx];
  for (int h = 0; h < NH_; ++h)
    bm[(size_t)h * (NQ * NKV) + idx] = rpb[r * NH_ + h];
}

// ---------------------------------------------------------------------------
// Generic bf16 WMMA GEMM:  out[M,N] = A[M,Kp] * W[N,Kp]^T + bias, epilogues:
//   mode 0: scatter kv -> kvh bf16 [M, 2*6*32]   (k|v, per-head pad to 32)
//   mode 1: scatter q  -> qh  bf16 [M, 6*32]     (* softmax scale)
//   mode 2: + x residual -> xo f32 [M,180]
//   mode 3: GELU -> h1 bf16 [M, 736]
//   mode 4: + xo residual -> d_out f32 [M,180]
// Block tile 64x128, 8 waves of 32x32 (2x2 WMMA tiles), K-step 32.
// Double-buffered LDS fed by global_load_async_to_lds_b128.
// ---------------------------------------------------------------------------
__global__ __launch_bounds__(256) void gemm_kernel(
    const u16* __restrict__ A, const u16* __restrict__ W,
    const float* __restrict__ bias, int N, int Kp, int mode, float scale,
    const float* __restrict__ add_f32, float* __restrict__ out_f32,
    u16* __restrict__ out_bf) {
  __shared__ u16 As[2][64 * 40];    // 80B row stride (16B aligned)
  __shared__ u16 Bs[2][128 * 40];

  int m0 = blockIdx.x * 64;
  int n0 = blockIdx.y * 128;
  int t = threadIdx.x;
  int wv = t >> 5, lane = t & 31;
  int wm = wv >> 2, wn = wv & 3;           // waves: 2 (M) x 4 (N)
  int lhalf = lane >> 4, lmod = lane & 15;

  v8f acc[2][2] = {};

  auto stage = [&](int buf, int k0) {
    { // A tile: 64 rows x 32 halves, one async b128 per thread
      int row = t >> 2, ch = t & 3;
      async_cp_b128(&As[buf][row * 40 + ch * 8],
                    A + (size_t)(m0 + row) * Kp + k0 + ch * 8);
    }
    for (int i = 0; i < 2; ++i) {  // B tile: 128 rows x 32 halves
      int cid = t * 2 + i;
      int row = cid >> 2, ch = cid & 3;
      int n = n0 + row;
      if (n < N) {
        async_cp_b128(&Bs[buf][row * 40 + ch * 8],
                      W + (size_t)n * Kp + k0 + ch * 8);
      } else {
        *(uint4*)&Bs[buf][row * 40 + ch * 8] = make_uint4(0, 0, 0, 0);
      }
    }
  };

  int nk = Kp >> 5;
  stage(0, 0);
  for (int ki = 0; ki < nk; ++ki) {
    int cur = ki & 1;
    wait_async0();            // this wave's tile(ki) landed in LDS
    __syncthreads();          // everyone's landed; prev readers of buf^1 done
    if (ki + 1 < nk) stage(cur ^ 1, (ki + 1) << 5);   // overlap with compute

    v16bf afrag[2], bfrag[2];
    for (int mi = 0; mi < 2; ++mi) {       // A 16x32 layout (ISA 7.12.2)
      int r = wm * 32 + mi * 16 + lmod;
      const u16* p = &As[cur][r * 40 + lhalf * 8];
      v16u tmp;
      ((uint4*)&tmp)[0] = *(const uint4*)p;          // K {0..7}/{8..15}
      ((uint4*)&tmp)[1] = *(const uint4*)(p + 16);   // K {16..23}/{24..31}
      afrag[mi] = __builtin_bit_cast(v16bf, tmp);
    }
    for (int ni = 0; ni < 2; ++ni) {       // B 32x16: lane=col, K {0..7,16..23}/{8..15,24..31}
      int r = wn * 32 + ni * 16 + lmod;
      const u16* p = &Bs[cur][r * 40 + lhalf * 8];
      v16u tmp;
      ((uint4*)&tmp)[0] = *(const uint4*)p;
      ((uint4*)&tmp)[1] = *(const uint4*)(p + 16);
      bfrag[ni] = __builtin_bit_cast(v16bf, tmp);
    }
    for (int mi = 0; mi < 2; ++mi)
      for (int ni = 0; ni < 2; ++ni)
        acc[mi][ni] = wmma_bf16(afrag[mi], bfrag[ni], acc[mi][ni]);
    __syncthreads();
  }

  // Epilogue. C/D layout: vgpr r -> row r (+8 for lanes 16..31), col = lane&15
  for (int mi = 0; mi < 2; ++mi)
    for (int ni = 0; ni < 2; ++ni)
      for (int r = 0; r < 8; ++r) {
        int row = m0 + wm * 32 + mi * 16 + r + ((lane >= 16) ? 8 : 0);
        int col = n0 + wn * 32 + ni * 16 + lmod;
        if (col >= N) continue;
        float v = acc[mi][ni][r] + bias[col];
        if (mode == 0) {
          int which = col / C_, cc = col % C_;
          int head = cc / D_, dd = cc % D_;
          out_bf[(size_t)row * (2 * NH_ * DP) + which * (NH_ * DP) + head * DP + dd] = f2bf(v);
        } else if (mode == 1) {
          int head = col / D_, dd = col % D_;
          out_bf[(size_t)row * (NH_ * DP) + head * DP + dd] = f2bf(v * scale);
        } else if (mode == 2 || mode == 4) {
          size_t o = (size_t)row * C_ + col;
          out_f32[o] = v + add_f32[o];
        } else { // mode 3: exact GELU
          float gg = 0.5f * v * (1.f + erff(v * 0.70710678118f));
          out_bf[(size_t)row * C4P + col] = f2bf(gg);
        }
      }
}

// ---------------------------------------------------------------------------
// Overlapping-window attention, one block per (window, head), 8 waves.
// Flash-style streaming softmax over 9 chunks of 64 keys; S never leaves LDS.
// LDS: k [576][32]bf16 (async-staged), vT [32][576]bf16, scratch [32][68]f32.
// ---------------------------------------------------------------------------
__global__ __launch_bounds__(256) void attn_kernel(
    const u16* __restrict__ qh,    // [M, 6*32] bf16, pre-scaled
    const u16* __restrict__ kvh,   // [M, 2*6*32] bf16
    const float* __restrict__ bm,  // [6][256][576] f32
    u16* __restrict__ out) {       // [M, 192] bf16, col = head*30 + d
  __shared__ u16  kbuf[NKV * DP];       // 36864 B
  __shared__ u16  vbufT[DP * NKV];      // 36864 B (transposed: [d][n])
  __shared__ float sbuf[8 * 32 * 68];   // 69632 B, per-wave score scratch
  __shared__ float redbuf[2][8 * 32];   // rescale + row-sum

  int blk = blockIdx.x;
  int head = blk % NH_;
  int w = blk / NH_;
  int b = w >> 4, wi = (w >> 2) & 3, wj = w & 3;
  int t = threadIdx.x, wv = t >> 5, lane = t & 31;
  int lhalf = lane >> 4, lmod = lane & 15;

  // ---- stage K rows (async) and transposed V (zero halo = spatial pad)
  for (int n = t; n < NKV; n += 256) {
    int dy = n / OWSZ, dx = n % OWSZ;
    int y = wi * WSZ - 4 + dy, x = wj * WSZ - 4 + dx;
    bool ok = ((unsigned)y < (unsigned)HW) && ((unsigned)x < (unsigned)HW);
    if (ok) {
      const u16* src = kvh + ((size_t)(b * L + y * HW + x)) * (2 * NH_ * DP) + head * DP;
      for (int c = 0; c < 4; ++c)
        async_cp_b128(&kbuf[n * DP + c * 8], src + c * 8);
      uint4 vr[4];
      const u16* vsrc = src + NH_ * DP;
      for (int c = 0; c < 4; ++c) vr[c] = *(const uint4*)(vsrc + c * 8);
      const u16* vvp = (const u16*)vr;
      for (int d = 0; d < DP; ++d) vbufT[d * NKV + n] = vvp[d];
    } else {
      for (int c = 0; c < 4; ++c)
        *(uint4*)&kbuf[n * DP + c * 8] = make_uint4(0, 0, 0, 0);
      for (int d = 0; d < DP; ++d) vbufT[d * NKV + n] = 0;
    }
  }
  wait_async0();
  __syncthreads();

  // ---- q A-fragments (wave owns query rows wv*32 .. wv*32+31)
  v16bf qf[2];
  for (int mi = 0; mi < 2; ++mi) {
    int qrow = wv * 32 + mi * 16 + lmod;               // row-major in window
    int qr = qrow >> 4, qc = qrow & 15;
    size_t grow = (size_t)b * L + (wi * WSZ + qr) * HW + (wj * WSZ + qc);
    const u16* p = qh + grow * (NH_ * DP) + head * DP + lhalf * 8;
    v16u tmp;
    ((uint4*)&tmp)[0] = *(const uint4*)p;
    ((uint4*)&tmp)[1] = *(const uint4*)(p + 16);
    qf[mi] = __builtin_bit_cast(v16bf, tmp);
  }

  float* srow_base = &sbuf[wv * 32 * 68];
  float* rbuf = &redbuf[0][wv * 32];
  float* lbuf = &redbuf[1][wv * 32];
  float m_run = -1e30f, l_run = 0.f;
  v8f o[2][2] = {};

  for (int c = 0; c < 9; ++c) {          // 9 chunks x 64 keys
    int nbase = c * 64;
    // ---- S = q * k^T  (4 N-tiles x 2 M-tiles, K=32 single WMMA each)
    for (int ni = 0; ni < 4; ++ni) {
      int kr = nbase + ni * 16 + lmod;
      const u16* p = &kbuf[kr * DP + lhalf * 8];
      v16u tmp;
      ((uint4*)&tmp)[0] = *(const uint4*)p;
      ((uint4*)&tmp)[1] = *(const uint4*)(p + 16);
      v16bf kf = __builtin_bit_cast(v16bf, tmp);
      for (int mi = 0; mi < 2; ++mi) {
        v8f z = {};
        v8f s = wmma_bf16(qf[mi], kf, z);
        for (int r = 0; r < 8; ++r) {
          int rr = mi * 16 + r + ((lane >= 16) ? 8 : 0);
          srow_base[rr * 68 + ni * 16 + lmod] = s[r];
        }
      }
    }
    // ---- streaming softmax, lane owns score row `lane` (wave-local LDS)
    {
      float* srow = srow_base + lane * 68;
      const float4* b4 = (const float4*)(
          bm + ((size_t)head * NQ + wv * 32 + lane) * NKV + nbase);
      float mx = -1e30f;
      for (int j4 = 0; j4 < 16; ++j4) {
        float4 sv = *(float4*)&srow[j4 * 4];
        float4 bv = b4[j4];
        sv.x += bv.x; sv.y += bv.y; sv.z += bv.z; sv.w += bv.w;
        *(float4*)&srow[j4 * 4] = sv;
        mx = fmaxf(mx, fmaxf(fmaxf(sv.x, sv.y), fmaxf(sv.z, sv.w)));
      }
      float newm = fmaxf(m_run, mx);
      float resc = __expf(m_run - newm);
      float ssum = 0.f;
      u16* prow = (u16*)srow;          // overlay: slot j/2 already consumed
      for (int j4 = 0; j4 < 16; ++j4) {
        float4 sv = *(float4*)&srow[j4 * 4];
        float p0 = __expf(sv.x - newm), p1 = __expf(sv.y - newm);
        float p2 = __expf(sv.z - newm), p3 = __expf(sv.w - newm);
        ssum += (p0 + p1) + (p2 + p3);
        uint2 pk;
        pk.x = (unsigned)f2bf(p0) | ((unsigned)f2bf(p1) << 16);
        pk.y = (unsigned)f2bf(p2) | ((unsigned)f2bf(p3) << 16);
        *(uint2*)&prow[j4 * 4] = pk;
      }
      l_run = l_run * resc + ssum;
      m_run = newm;
      rbuf[lane] = resc;
    }
    // ---- rescale O accumulators by the owning row's factor
    for (int mi = 0; mi < 2; ++mi)
      for (int dj = 0; dj < 2; ++dj)
        for (int r = 0; r < 8; ++r) {
          int rr = mi * 16 + r + ((lane >= 16) ? 8 : 0);
          o[mi][dj][r] *= rbuf[rr];
        }
    // ---- O += P * V   (chunk K=64 -> 2 WMMA K-steps)
    for (int ks = 0; ks < 2; ++ks) {
      v16bf pf[2], vf[2];
      for (int mi = 0; mi < 2; ++mi) {
        int ml = mi * 16 + lmod;
        const u16* p = (const u16*)(srow_base + ml * 68) + ks * 32 + lhalf * 8;
        v16u tmp;
        ((uint4*)&tmp)[0] = *(const uint4*)p;
        ((uint4*)&tmp)[1] = *(const uint4*)(p + 16);
        pf[mi] = __builtin_bit_cast(v16bf, tmp);
      }
      for (int dj = 0; dj < 2; ++dj) {
        int d = dj * 16 + lmod;
        const u16* p = &vbufT[d * NKV + nbase + ks * 32 + lhalf * 8];
        v16u tmp;
        ((uint4*)&tmp)[0] = *(const uint4*)p;
        ((uint4*)&tmp)[1] = *(const uint4*)(p + 16);
        vf[dj] = __builtin_bit_cast(v16bf, tmp);
      }
      for (int mi = 0; mi < 2; ++mi)
        for (int dj = 0; dj < 2; ++dj)
          o[mi][dj] = wmma_bf16(pf[mi], vf[dj], o[mi][dj]);
    }
  }

  lbuf[lane] = l_run;
  // ---- normalize and scatter back to spatial layout (window_reverse fused)
  for (int mi = 0; mi < 2; ++mi)
    for (int dj = 0; dj < 2; ++dj)
      for (int r = 0; r < 8; ++r) {
        int rr = mi * 16 + r + ((lane >= 16) ? 8 : 0);
        float val = o[mi][dj][r] / lbuf[rr];
        int d = dj * 16 + lmod;
        if (d < D_) {
          int qrow = wv * 32 + rr;
          int qr = qrow >> 4, qc = qrow & 15;
          size_t grow = (size_t)b * L + (wi * WSZ + qr) * HW + (wj * WSZ + qc);
          out[grow * CP + head * D_ + d] = f2bf(val);
        }
      }
}

// ---------------------------------------------------------------------------
extern "C" void kernel_launch(void* const* d_in, const int* in_sizes, int n_in,
                              void* d_out, int out_size, void* d_ws, size_t ws_size,
                              hipStream_t stream) {
  const float* x      = (const float*)d_in[0];
  const float* y      = (const float*)d_in[1];
  const float* n1g    = (const float*)d_in[2];
  const float* n1b    = (const float*)d_in[3];
  const float* kv_w   = (const float*)d_in[4];
  const float* kv_b   = (const float*)d_in[5];
  const float* q_w    = (const float*)d_in[6];
  const float* q_b    = (const float*)d_in[7];
  const float* rpb    = (const float*)d_in[8];
  const float* proj_w = (const float*)d_in[9];
  const float* proj_b = (const float*)d_in[10];
  const float* n2g    = (const float*)d_in[11];
  const float* n2b    = (const float*)d_in[12];
  const float* fc1_w  = (const float*)d_in[13];
  const float* fc1_b  = (const float*)d_in[14];
  const float* fc2_w  = (const float*)d_in[15];
  const float* fc2_b  = (const float*)d_in[16];
  const int*   rpi    = (const int*)d_in[17];
  // d_in[18]/[19] = h,w fixed at 64

  // workspace layout (~85 MB)
  char* p = (char*)d_ws;
  auto take = [&](size_t bytes) { char* r = p; p += (bytes + 255) & ~255ull; return r; };
  u16*   xnbuf   = (u16*)  take((size_t)M_ROWS * CP  * 2);
  u16*   ynbuf   = (u16*)  take((size_t)M_ROWS * CP  * 2);
  u16*   kvwb    = (u16*)  take((size_t)360 * CP * 2);
  u16*   qwb     = (u16*)  take((size_t)C_  * CP * 2);
  u16*   projwb  = (u16*)  take((size_t)C_  * CP * 2);
  u16*   fc1wb   = (u16*)  take((size_t)C4  * CP * 2);
  u16*   fc2wb   = (u16*)  take((size_t)C_  * C4P * 2);
  u16*   kvh     = (u16*)  take((size_t)M_ROWS * 2 * NH_ * DP * 2);
  u16*   qh      = (u16*)  take((size_t)M_ROWS * NH_ * DP * 2);
  u16*   attn_o  = (u16*)  take((size_t)M_ROWS * CP * 2);
  float* xo      = (float*)take((size_t)M_ROWS * C_ * 4);
  u16*   xn2buf  = (u16*)  take((size_t)M_ROWS * CP * 2);
  u16*   h1      = (u16*)  take((size_t)M_ROWS * C4P * 2);
  float* biasmat = (float*)take((size_t)NH_ * NQ * NKV * 4);

  // q pads (per-head cols 30,31) must be zero so garbage k pads cancel in S
  hipMemsetAsync(qh, 0, (size_t)M_ROWS * NH_ * DP * 2, stream);

  ln_cvt_kernel<<<M_ROWS / 8, 256, 0, stream>>>(x, n1g, n1b, xnbuf, M_ROWS);
  ln_cvt_kernel<<<M_ROWS / 8, 256, 0, stream>>>(y, n1g, n1b, ynbuf, M_ROWS);

  cvt_w_kernel<<<(360 * CP + 255) / 256, 256, 0, stream>>>(kv_w,   kvwb,   360, C_,  CP);
  cvt_w_kernel<<<(C_  * CP + 255) / 256, 256, 0, stream>>>(q_w,    qwb,    C_,  C_,  CP);
  cvt_w_kernel<<<(C_  * CP + 255) / 256, 256, 0, stream>>>(proj_w, projwb, C_,  C_,  CP);
  cvt_w_kernel<<<(C4  * CP + 255) / 256, 256, 0, stream>>>(fc1_w,  fc1wb,  C4,  C_,  CP);
  cvt_w_kernel<<<(C_  * C4P + 255) / 256, 256, 0, stream>>>(fc2_w, fc2wb,  C_,  C4,  C4P);

  bias_expand_kernel<<<(NQ * NKV) / 256, 256, 0, stream>>>(rpb, rpi, biasmat);

  const float qscale = 0.18257418583505537f;   // (30)^-0.5
  // kv = xn @ kv_w^T + kv_b   -> head-padded k|v
  gemm_kernel<<<dim3(M_ROWS / 64, 3), 256, 0, stream>>>(
      xnbuf, kvwb, kv_b, 360, CP, 0, 1.f, nullptr, nullptr, kvh);
  // q = (yn @ q_w^T + q_b) * scale
  gemm_kernel<<<dim3(M_ROWS / 64, 2), 256, 0, stream>>>(
      ynbuf, qwb, q_b, C_, CP, 1, qscale, nullptr, nullptr, qh);

  attn_kernel<<<64 * NH_, 256, 0, stream>>>(qh, kvh, biasmat, attn_o);

  // xo = attn @ proj_w^T + proj_b + x
  gemm_kernel<<<dim3(M_ROWS / 64, 2), 256, 0, stream>>>(
      attn_o, projwb, proj_b, C_, CP, 2, 1.f, x, xo, nullptr);

  ln_cvt_kernel<<<M_ROWS / 8, 256, 0, stream>>>(xo, n2g, n2b, xn2buf, M_ROWS);

  // h1 = gelu(xn2 @ fc1_w^T + fc1_b)
  gemm_kernel<<<dim3(M_ROWS / 64, 6), 256, 0, stream>>>(
      xn2buf, fc1wb, fc1_b, C4, CP, 3, 1.f, nullptr, nullptr, h1);
  // out = h1 @ fc2_w^T + fc2_b + xo
  gemm_kernel<<<dim3(M_ROWS / 64, 2), 256, 0, stream>>>(
      h1, fc2wb, fc2_b, C_, C4P, 4, 1.f, xo, (float*)d_out, nullptr);
}
